// _ConvBNReLU_85409719648748
// MI455X (gfx1250) — matrix operand order repaired and checked by reading, not compiled, not run on hardware
//
#include <hip/hip_runtime.h>

typedef __attribute__((ext_vector_type(16))) _Float16 v16h;
typedef __attribute__((ext_vector_type(8)))  _Float16 v8h;
typedef __attribute__((ext_vector_type(8)))  float    v8f;

#define C_IN   64
#define C_OUT  128
#define HW     128
#define N_IMG  16
#define WSZ    (C_OUT * C_IN * 9)     // 73728 weight elements
#define PXSTRIDE 72                   // halfs per pixel row in LDS (144B: 16B-aligned, conflict-friendly)

// d_ws layout:
//   float[0]        : w_scale (= max|W| / 127)
//   float[16..143]  : per-channel BN scale
//   float[144..271] : per-channel BN shift
//   byte 2048 ...   : _Float16 Wt[9][128][64]   (quant-dequantized weights, [r*3+s][o][c])

// ---------------- prep 1: w_scale = max|W| / 127 ----------------
__global__ void k_wscale(const float* __restrict__ W, float* __restrict__ ws_f) {
    __shared__ float red[256];
    float m = 0.0f;
    for (int i = threadIdx.x; i < WSZ; i += 256) m = fmaxf(m, fabsf(W[i]));
    red[threadIdx.x] = m;
    __syncthreads();
    for (int off = 128; off > 0; off >>= 1) {
        if ((int)threadIdx.x < off)
            red[threadIdx.x] = fmaxf(red[threadIdx.x], red[threadIdx.x + off]);
        __syncthreads();
    }
    if (threadIdx.x == 0) ws_f[0] = red[0] * (1.0f / 127.0f);
}

// ---------------- prep 2: quant-dequant W -> f16 [rs][o][c]; BN scale/shift ----------------
__global__ void k_prep(const float* __restrict__ W,
                       const float* __restrict__ gamma, const float* __restrict__ beta,
                       const float* __restrict__ mean,  const float* __restrict__ var,
                       float* __restrict__ ws_f, _Float16* __restrict__ Wt) {
    int idx = blockIdx.x * 256 + threadIdx.x;
    float wsc = ws_f[0];
    if (idx < WSZ) {
        // W is OIHW: idx = ((o*64 + c)*3 + r)*3 + s
        int s = idx % 3;
        int r = (idx / 3) % 3;
        int c = (idx / 9) % C_IN;
        int o = idx / (9 * C_IN);
        float q = rintf(W[idx] / wsc);               // round-half-even (RNE default)
        q = fminf(fmaxf(q, -127.0f), 127.0f) * wsc;  // dequant
        Wt[(((r * 3 + s) * C_OUT + o) * C_IN) + c] = (_Float16)q;
    }
    if (idx < C_OUT) {
        float inv = 1.0f / sqrtf(var[idx] + 1e-5f);
        float sc  = gamma[idx] * inv;
        ws_f[16 + idx]  = sc;
        ws_f[144 + idx] = beta[idx] - mean[idx] * sc;
    }
}

// ---------------- main: implicit-GEMM conv via v_wmma_f32_16x16x32_f16 ----------------
// Block = one (n, h): 128 out-channels x 128 pixels. 8 waves; wave w owns
// channels [16w, 16w+16) x all 128 pixels -> 8 accumulators of v8f.
// Per (kc,s) group: batch-load ALL 8 B-fragments (16 ds_load_b128 into 64
// distinct VGPRs) before the 8-WMMA burst, so LDS latency overlaps the WMMAs
// instead of a full s_wait_dscnt drain per WMMA. Next group's weight fragment
// is prefetched behind the WMMA burst.
__global__ void __launch_bounds__(256)
k_conv(const float* __restrict__ x, const _Float16* __restrict__ Wt,
       const float* __restrict__ ws_f, const float* __restrict__ act_scale,
       float* __restrict__ out) {
    __shared__ _Float16 slab[130 * PXSTRIDE];   // [px 0..129][c 0..63]; px 0 <=> w = -1

    const int h    = blockIdx.x;
    const int n    = blockIdx.y;
    const int t    = threadIdx.x;
    const int wave = t >> 5;
    const int lane = t & 31;
    const int lhalf = lane >> 4;     // which 16-lane half
    const int l16   = lane & 15;
    const int chbase = wave * 16;

    v8f acc[8];
#pragma unroll
    for (int i = 0; i < 8; ++i) acc[i] = (v8f){0,0,0,0,0,0,0,0};

    // per-lane fragment base pointer (all step offsets become immediates)
    const _Float16* bbase = &slab[l16 * PXSTRIDE + 16 * lhalf];

    for (int r = 0; r < 3; ++r) {
        const int row = h + r - 1;
        if ((unsigned)row >= (unsigned)HW) continue;   // block-uniform zero-padding skip

        // ---- stage x[n, :, row, -1..128] into LDS as f16, [pixel][channel] ----
        for (int idx = t; idx < C_IN * 130; idx += 256) {
            int c  = idx / 130;
            int px = idx - c * 130;          // consecutive threads -> consecutive pixels (coalesced)
            int wx = px - 1;
            float v = 0.0f;
            if ((unsigned)wx < (unsigned)HW)
                v = x[(((size_t)n * C_IN + c) * HW + row) * HW + wx];
            slab[px * PXSTRIDE + c] = (_Float16)v;
        }
        __syncthreads();

        const _Float16* wbase =
            Wt + ((size_t)(r * 3) * C_OUT + chbase + l16) * C_IN + 16 * lhalf;

        // A fragment (weights, 16x32): lane holds M=l16, K = 16*lhalf + 0..15
        auto ldA = [&](int ks) -> v16h {   // ks = kc*3... flattened group: kc=ks/3, s=ks%3
            const int kc = ks / 3, s = ks % 3;
            const _Float16* p = wbase + (size_t)s * (C_OUT * C_IN) + kc * 32;
            v8h lo = *(const v8h*)p;          // global_load_b128
            v8h hi = *(const v8h*)(p + 8);    // global_load_b128
            return __builtin_shufflevector(lo, hi,
                0,1,2,3,4,5,6,7,8,9,10,11,12,13,14,15);
        };
        // B fragment (activations, 32x16): lane N = pixel l16, K = 16*lhalf + 0..15
        auto ldB = [&](int ks, int pt) -> v16h {
            const int kc = ks / 3, s = ks % 3;
            const _Float16* p = bbase + (pt * 16 + s) * PXSTRIDE + kc * 32;
            v8h lo = *(const v8h*)p;          // ds_load_b128
            v8h hi = *(const v8h*)(p + 8);    // ds_load_b128
            return __builtin_shufflevector(lo, hi,
                0,1,2,3,4,5,6,7,8,9,10,11,12,13,14,15);
        };

        v16h awc = ldA(0);
#pragma unroll
        for (int ks = 0; ks < 6; ++ks) {     // 2 kc-chunks x 3 s-taps
            // batch-load all 8 B fragments of this group: 64 distinct VGPRs,
            // 16 ds_load_b128 issued before the first WMMA consumer
            v16h bf[8];
#pragma unroll
            for (int pt = 0; pt < 8; ++pt) bf[pt] = ldB(ks, pt);

            // prefetch next group's weight fragment behind the WMMA burst
            v16h awn;
            if (ks < 5) awn = ldA(ks + 1);

#pragma unroll
            for (int pt = 0; pt < 8; ++pt)
                acc[pt] = __builtin_amdgcn_wmma_f32_16x16x32_f16(
                    false, awc, false, bf[pt], (short)0, acc[pt], false, false);

            if (ks < 5) awc = awn;
        }
        __syncthreads();
    }

    // ---- epilogue: BN + ReLU + uint8 quant-dequant; coalesced stores (N dim = pixels) ----
    const float sact = act_scale[0] * (1.0f / 255.0f);
    const float inv_sact = 1.0f / sact;
    float sc[8], sh[8];
#pragma unroll
    for (int v = 0; v < 8; ++v) {
        int ch = chbase + v + 8 * lhalf;   // D layout: lane -> N = l16 (pixel), VGPR v -> M = v + 8*lhalf (channel)
        sc[v] = ws_f[16 + ch];
        sh[v] = ws_f[144 + ch];
    }
#pragma unroll
    for (int pt = 0; pt < 8; ++pt) {
        int px = pt * 16 + l16;
#pragma unroll
        for (int v = 0; v < 8; ++v) {
            int ch = chbase + v + 8 * lhalf;
            float y = acc[pt][v] * sc[v] + sh[v];
            y = fmaxf(y, 0.0f);                                  // ReLU
            float q = fminf(rintf(y * inv_sact), 255.0f) * sact; // uint8 QDQ (>=0 after ReLU)
            out[(((size_t)n * C_OUT + ch) * HW + h) * HW + px] = q;
        }
    }
}

extern "C" void kernel_launch(void* const* d_in, const int* in_sizes, int n_in,
                              void* d_out, int out_size, void* d_ws, size_t ws_size,
                              hipStream_t stream) {
    const float* x     = (const float*)d_in[0];
    const float* W     = (const float*)d_in[1];
    const float* gamma = (const float*)d_in[2];
    const float* beta  = (const float*)d_in[3];
    const float* mean  = (const float*)d_in[4];
    const float* var   = (const float*)d_in[5];
    const float* ascl  = (const float*)d_in[6];
    float* out = (float*)d_out;

    float*     ws_f = (float*)d_ws;
    _Float16*  Wt   = (_Float16*)((char*)d_ws + 2048);

    k_wscale<<<1, 256, 0, stream>>>(W, ws_f);
    k_prep<<<(WSZ + 255) / 256, 256, 0, stream>>>(W, gamma, beta, mean, var, ws_f, Wt);
    k_conv<<<dim3(HW, N_IMG), 256, 0, stream>>>(x, Wt, ws_f, ascl, out);
}
